// MambaSSD_8211977470176
// MI455X (gfx1250) — matrix operand order repaired and checked by reading, not compiled, not run on hardware
//
#include <hip/hip_runtime.h>
#include <hip/hip_bf16.h>
#include <math.h>

// ---------------------------------------------------------------------------
// Problem constants (from reference)
// ---------------------------------------------------------------------------
#define D_MODEL   1024
#define D_STATE   64
#define D_INNER   2048          // EXPAND * D_MODEL
#define BATCH     4
#define SEQ       2048
#define ROWS      (BATCH * SEQ) // 8192 "token" rows
#define N_XP_PAD  160           // 2*D_STATE+1 = 129 padded to multiple of 32
#define NCHUNK    16            // scan chunks per sequence
#define CHUNK     (SEQ / NCHUNK)

typedef __bf16 bf16_t;
typedef __attribute__((ext_vector_type(16))) __bf16 v16bf;
typedef __attribute__((ext_vector_type(8)))  __bf16 v8bf;
typedef __attribute__((ext_vector_type(8)))  float  v8f;

// ---------------------------------------------------------------------------
// Elementwise helpers
// ---------------------------------------------------------------------------
__global__ __launch_bounds__(256) void k_f32_to_bf16(const float* __restrict__ in,
                                                     bf16_t* __restrict__ out, int n) {
    int i = blockIdx.x * 256 + threadIdx.x;
    if (i < n) out[i] = (bf16_t)in[i];
}

// out[N_out, K] = transpose(in[K, N_real]) in bf16, rows >= N_real zero-filled.
__global__ __launch_bounds__(256) void k_transpose_bf16(const float* __restrict__ in,
                                                        bf16_t* __restrict__ out,
                                                        int K, int Nreal, int Nout) {
    int i = blockIdx.x * 256 + threadIdx.x;
    if (i >= Nout * K) return;
    int n = i / K;
    int k = i - n * K;
    out[i] = (n < Nreal) ? (bf16_t)in[(size_t)k * Nreal + n] : (bf16_t)0.f;
}

// From proj[ROWS, N_XP_PAD] (fp32): abar[i] = exp(-exp(A_log) * softplus(proj[i,0])),
// Braw/Craw bf16 [ROWS, 64].
__global__ __launch_bounds__(256) void k_prep_scan(const float* __restrict__ proj,
                                                   const float* __restrict__ A_log,
                                                   float* __restrict__ abar,
                                                   bf16_t* __restrict__ Braw,
                                                   bf16_t* __restrict__ Craw) {
    int tid = blockIdx.x * 256 + threadIdx.x;
    if (tid >= ROWS * D_STATE) return;
    int i = tid >> 6;          // row
    int j = tid & 63;          // state index
    const float* row = proj + (size_t)i * N_XP_PAD;
    Braw[(size_t)i * D_STATE + j] = (bf16_t)row[1 + j];
    Craw[(size_t)i * D_STATE + j] = (bf16_t)row[1 + D_STATE + j];
    if (j == 0) {
        float p = row[0];
        float delta = (p > 20.f) ? p : log1pf(__expf(p));
        float eA = __expf(A_log[0]);            // A = -exp(A_log)
        abar[i] = __expf(-eA * delta);          // exp(delta * A)
    }
}

// Per-chunk scalar products of abar: Ptot[b*NCHUNK+c] = prod_{t in chunk} a_t
__global__ __launch_bounds__(64) void k_chunkprod(const float* __restrict__ abar,
                                                  float* __restrict__ Ptot) {
    int i = threadIdx.x;
    if (i >= BATCH * NCHUNK) return;
    int b = i >> 4, c = i & (NCHUNK - 1);
    const float* ap = abar + b * SEQ + c * CHUNK;
    float p = 1.f;
    for (int t = 0; t < CHUNK; ++t) p *= ap[t];
    Ptot[i] = p;
}

// Pass 1: local scan per chunk (h0 = 0), emit chunk-end carry hc[b,c,d].
__global__ __launch_bounds__(256) void k_scan_pass1(const bf16_t* __restrict__ Bb,
                                                    const bf16_t* __restrict__ xib,
                                                    const float*  __restrict__ abar,
                                                    float* __restrict__ hc) {
    int d = blockIdx.x * 256 + threadIdx.x;
    int c = blockIdx.y;
    int b = blockIdx.z;
    int base = b * SEQ + c * CHUNK;
    float h = 0.f;
    for (int t = 0; t < CHUNK; ++t) {
        size_t idx = (size_t)(base + t) * D_INNER + d;
        h = fmaf(abar[base + t], h, (float)Bb[idx] * (float)xib[idx]);
    }
    hc[((size_t)(b * NCHUNK + c)) * D_INNER + d] = h;
}

// Pass 2: reconstruct carry-in via 16-step prefix, recompute local scan, gate output.
__global__ __launch_bounds__(256) void k_scan_pass2(const bf16_t* __restrict__ Bb,
                                                    const bf16_t* __restrict__ Cb,
                                                    const bf16_t* __restrict__ xib,
                                                    const bf16_t* __restrict__ szb,
                                                    const float*  __restrict__ abar,
                                                    const float*  __restrict__ Ptot,
                                                    const float*  __restrict__ hc,
                                                    const float*  __restrict__ Dvec,
                                                    bf16_t* __restrict__ yb) {
    int d = blockIdx.x * 256 + threadIdx.x;
    int c = blockIdx.y;
    int b = blockIdx.z;
    float h = 0.f;
    for (int cc = 0; cc < c; ++cc)
        h = fmaf(Ptot[b * NCHUNK + cc], h, hc[((size_t)(b * NCHUNK + cc)) * D_INNER + d]);
    float Dv = Dvec[d];
    int base = b * SEQ + c * CHUNK;
    for (int t = 0; t < CHUNK; ++t) {
        size_t idx = (size_t)(base + t) * D_INNER + d;
        float a  = abar[base + t];
        float xi = (float)xib[idx];
        h = fmaf(a, h, (float)Bb[idx] * xi);
        float y = fmaf((float)Cb[idx], h, Dv * xi);
        y *= (float)szb[idx];
        yb[idx] = (bf16_t)y;
    }
}

// ---------------------------------------------------------------------------
// bf16 WMMA GEMM, 2x2 register-blocked: each wave computes a 32x32 output tile
// (4 accumulators) from A[M,K] and pre-transposed Bt[N,K], K-step 32 via
// v_wmma_f32_16x16x32_bf16. Per-lane fragment layouts used here (contiguous k, not the ISA table order):
//   A (16x32 bf16): lane<16 -> row=l16, K 0..15 contiguous; lane>=16 -> K 16..31
//   B (32x16 bf16): lane<16 -> col=l16, K 0..15 contiguous; lane>=16 -> K 16..31
//   C/D (16x16 f32): elem j -> row = half*8 + j, col = l16
// MODE 0: fp32 out (ld=ldo)  MODE 1: bf16 out (ld=N)
// MODE 2: split epilogue (cols<N/2 -> xi bf16, cols>=N/2 -> silu(z) bf16)
// ---------------------------------------------------------------------------
template <int MODE>
__global__ __launch_bounds__(256)
void gemm_bf16_wmma(const bf16_t* __restrict__ A, const bf16_t* __restrict__ Bt,
                    float* __restrict__ outF, bf16_t* __restrict__ outB0,
                    bf16_t* __restrict__ outB1, int M, int N, int K, int ldo) {
    const int lane = threadIdx.x & 31;
    const int wave = threadIdx.x >> 5;
    const int ntn  = N >> 5;                    // 32-wide tiles along N
    int tile = blockIdx.x * 8 + wave;
    int tn = tile % ntn;
    int tm = tile / ntn;
    if (tm * 32 >= M) return;

    const int l16  = lane & 15;
    const int half = lane >> 4;

    const bf16_t* Ap0 = A + (size_t)(tm * 32 + l16) * K + (half ? 16 : 0);
    const bf16_t* Ap1 = A + (size_t)(tm * 32 + 16 + l16) * K + (half ? 16 : 0);
    const bf16_t* Bp0 = Bt + (size_t)(tn * 32 + l16) * K + (half ? 16 : 0);
    const bf16_t* Bp1 = Bt + (size_t)(tn * 32 + 16 + l16) * K + (half ? 16 : 0);

    v8f acc00 = {}, acc01 = {}, acc10 = {}, acc11 = {};
    for (int k0 = 0; k0 < K; k0 += 32) {
        __builtin_prefetch((const void*)(Bp0 + k0 + 512), 0, 0);
        __builtin_prefetch((const void*)(Bp1 + k0 + 512), 0, 0);
        union U { v16bf v; v8bf h[2]; };
        U a0, a1;
        a0.h[0] = *(const v8bf*)(Ap0 + k0);
        a0.h[1] = *(const v8bf*)(Ap0 + k0 + 8);
        a1.h[0] = *(const v8bf*)(Ap1 + k0);
        a1.h[1] = *(const v8bf*)(Ap1 + k0 + 8);
        v16bf b0 = *(const v16bf*)(Bp0 + k0);
        v16bf b1 = *(const v16bf*)(Bp1 + k0);
        acc00 = __builtin_amdgcn_wmma_f32_16x16x32_bf16(false, a0.v, false, b0,
                                                        (short)0, acc00, false, false);
        acc01 = __builtin_amdgcn_wmma_f32_16x16x32_bf16(false, a0.v, false, b1,
                                                        (short)0, acc01, false, false);
        acc10 = __builtin_amdgcn_wmma_f32_16x16x32_bf16(false, a1.v, false, b0,
                                                        (short)0, acc10, false, false);
        acc11 = __builtin_amdgcn_wmma_f32_16x16x32_bf16(false, a1.v, false, b1,
                                                        (short)0, acc11, false, false);
    }

    int row0 = tm * 32 + half * 8;              // rows for the upper A-tile
    int col0 = tn * 32 + l16;                   // cols for the left  B-tile

    if (MODE == 0) {
        auto emit = [&](const v8f& a, int r, int cq) {
            #pragma unroll
            for (int j = 0; j < 8; ++j) outF[(size_t)(r + j) * ldo + cq] = a[j];
        };
        emit(acc00, row0, col0);       emit(acc01, row0, col0 + 16);
        emit(acc10, row0 + 16, col0);  emit(acc11, row0 + 16, col0 + 16);
    } else if (MODE == 1) {
        auto emit = [&](const v8f& a, int r, int cq) {
            #pragma unroll
            for (int j = 0; j < 8; ++j) outB0[(size_t)(r + j) * N + cq] = (bf16_t)a[j];
        };
        emit(acc00, row0, col0);       emit(acc01, row0, col0 + 16);
        emit(acc10, row0 + 16, col0);  emit(acc11, row0 + 16, col0 + 16);
    } else {
        const int Nh = N >> 1;
        const bool isXi = (tn * 32) < Nh;       // whole 32-col tile on one side
        bf16_t* dst = isXi ? outB0 : outB1;
        const int cadj = isXi ? 0 : Nh;
        auto emit = [&](const v8f& a, int r, int cq) {
            #pragma unroll
            for (int j = 0; j < 8; ++j) {
                float v = a[j];
                if (!isXi) v = v / (1.f + __expf(-v));     // silu for the z half
                dst[(size_t)(r + j) * Nh + (cq - cadj)] = (bf16_t)v;
            }
        };
        emit(acc00, row0, col0);       emit(acc01, row0, col0 + 16);
        emit(acc10, row0 + 16, col0);  emit(acc11, row0 + 16, col0 + 16);
    }
}

// ---------------------------------------------------------------------------
// Host-side orchestration
// ---------------------------------------------------------------------------
static inline size_t align256(size_t x) { return (x + 255) & ~(size_t)255; }

extern "C" void kernel_launch(void* const* d_in, const int* in_sizes, int n_in,
                              void* d_out, int out_size, void* d_ws, size_t ws_size,
                              hipStream_t stream) {
    (void)in_sizes; (void)n_in; (void)out_size; (void)ws_size;
    const float* x     = (const float*)d_in[0];  // [B,S,D_MODEL]
    const float* W_in  = (const float*)d_in[1];  // [D_MODEL, 2*D_INNER]
    const float* W_xp  = (const float*)d_in[2];  // [D_INNER, 129]
    const float* W_B   = (const float*)d_in[3];  // [D_STATE, D_INNER]
    const float* W_C   = (const float*)d_in[4];  // [D_STATE, D_INNER]
    const float* W_out = (const float*)d_in[5];  // [D_INNER, D_MODEL]
    const float* Dvec  = (const float*)d_in[6];  // [D_INNER]
    const float* A_log = (const float*)d_in[7];  // scalar
    float* out = (float*)d_out;                  // [B,S,D_MODEL]

    // ---- workspace carve-up (bytes) ----
    char* ws = (char*)d_ws;
    size_t off = 0;
    auto take = [&](size_t bytes) { char* p = ws + off; off += align256(bytes); return p; };

    bf16_t* xb    = (bf16_t*)take((size_t)ROWS * D_MODEL * 2);          // x bf16
    bf16_t* WinT  = (bf16_t*)take((size_t)(2 * D_INNER) * D_MODEL * 2); // [4096,1024]
    bf16_t* WxpT  = (bf16_t*)take((size_t)N_XP_PAD * D_INNER * 2);      // [160,2048]
    bf16_t* WBT   = (bf16_t*)take((size_t)D_INNER * D_STATE * 2);       // [2048,64]
    bf16_t* WCT   = (bf16_t*)take((size_t)D_INNER * D_STATE * 2);       // [2048,64]
    bf16_t* WoutT = (bf16_t*)take((size_t)D_MODEL * D_INNER * 2);       // [1024,2048]
    bf16_t* xib   = (bf16_t*)take((size_t)ROWS * D_INNER * 2);          // xi bf16
    bf16_t* szb   = (bf16_t*)take((size_t)ROWS * D_INNER * 2);          // silu(z) bf16
    float*  proj  = (float*) take((size_t)ROWS * N_XP_PAD * 4);         // fp32, ld=160
    float*  abar  = (float*) take((size_t)ROWS * 4);
    bf16_t* Braw  = (bf16_t*)take((size_t)ROWS * D_STATE * 2);
    bf16_t* Craw  = (bf16_t*)take((size_t)ROWS * D_STATE * 2);
    bf16_t* Bb    = (bf16_t*)take((size_t)ROWS * D_INNER * 2);
    bf16_t* Cb    = (bf16_t*)take((size_t)ROWS * D_INNER * 2);
    bf16_t* yb    = (bf16_t*)take((size_t)ROWS * D_INNER * 2);
    float*  hc    = (float*) take((size_t)BATCH * NCHUNK * D_INNER * 4);
    float*  Ptot  = (float*) take((size_t)BATCH * NCHUNK * 4);

    // ---- 1) precision conversion / weight transposes ----
    {
        int n = ROWS * D_MODEL;
        k_f32_to_bf16<<<(n + 255) / 256, 256, 0, stream>>>(x, xb, n);
    }
    { int n = (2 * D_INNER) * D_MODEL;
      k_transpose_bf16<<<(n + 255) / 256, 256, 0, stream>>>(W_in, WinT, D_MODEL, 2 * D_INNER, 2 * D_INNER); }
    { int n = N_XP_PAD * D_INNER;
      k_transpose_bf16<<<(n + 255) / 256, 256, 0, stream>>>(W_xp, WxpT, D_INNER, 2 * D_STATE + 1, N_XP_PAD); }
    { int n = D_INNER * D_STATE;
      k_transpose_bf16<<<(n + 255) / 256, 256, 0, stream>>>(W_B, WBT, D_STATE, D_INNER, D_INNER); }
    { int n = D_INNER * D_STATE;
      k_transpose_bf16<<<(n + 255) / 256, 256, 0, stream>>>(W_C, WCT, D_STATE, D_INNER, D_INNER); }
    { int n = D_MODEL * D_INNER;
      k_transpose_bf16<<<(n + 255) / 256, 256, 0, stream>>>(W_out, WoutT, D_INNER, D_MODEL, D_MODEL); }

    // ---- 2) GEMM1: xz = x @ W_in, fused split -> xi (bf16) and silu(z) (bf16) ----
    {
        int tiles = (ROWS / 32) * ((2 * D_INNER) / 32);   // 32768
        gemm_bf16_wmma<2><<<tiles / 8, 256, 0, stream>>>(
            xb, WinT, nullptr, xib, szb, ROWS, 2 * D_INNER, D_MODEL, 0);
    }

    // ---- 3) GEMM2: proj = xi @ W_xp (N padded to 160), fp32 out ----
    {
        int tiles = (ROWS / 32) * (N_XP_PAD / 32);        // 1280
        gemm_bf16_wmma<0><<<tiles / 8, 256, 0, stream>>>(
            xib, WxpT, proj, nullptr, nullptr, ROWS, N_XP_PAD, D_INNER, N_XP_PAD);
    }

    // ---- 4) delta/softplus/abar + extract B_raw, C_raw ----
    {
        int n = ROWS * D_STATE;
        k_prep_scan<<<(n + 255) / 256, 256, 0, stream>>>(proj, A_log, abar, Braw, Craw);
    }

    // ---- 5) GEMM3/4: B = Braw @ W_B, C = Craw @ W_C (K = 64), bf16 out ----
    {
        int tiles = (ROWS / 32) * (D_INNER / 32);         // 16384
        gemm_bf16_wmma<1><<<tiles / 8, 256, 0, stream>>>(
            Braw, WBT, nullptr, Bb, nullptr, ROWS, D_INNER, D_STATE, 0);
        gemm_bf16_wmma<1><<<tiles / 8, 256, 0, stream>>>(
            Craw, WCT, nullptr, Cb, nullptr, ROWS, D_INNER, D_STATE, 0);
    }

    // ---- 6) chunked two-pass recurrence scan (fp32 state in registers) ----
    {
        k_chunkprod<<<1, 64, 0, stream>>>(abar, Ptot);
        dim3 grid(D_INNER / 256, NCHUNK, BATCH);          // 512 blocks / pass
        k_scan_pass1<<<grid, 256, 0, stream>>>(Bb, xib, abar, hc);
        k_scan_pass2<<<grid, 256, 0, stream>>>(Bb, Cb, xib, szb, abar, Ptot, hc, Dvec, yb);
    }

    // ---- 7) GEMM5: out = y @ W_out, fp32 out ----
    {
        int tiles = (ROWS / 32) * (D_MODEL / 32);         // 8192
        gemm_bf16_wmma<0><<<tiles / 8, 256, 0, stream>>>(
            yb, WoutT, out, nullptr, nullptr, ROWS, D_MODEL, D_INNER, D_MODEL);
    }
}
